// HebbianEmbedding_37151467110560
// MI455X (gfx1250) — matrix operand order repaired and checked by reading, not compiled, not run on hardware
//
#include <hip/hip_runtime.h>
#include <math.h>

#define VOCAB   50257
#define EMBED   768
#define SEQ     1024
#define MIN_TOK 5
#define EPSF    1e-12f

typedef float v2f  __attribute__((ext_vector_type(2)));
typedef float v4f  __attribute__((ext_vector_type(4)));
typedef float v8f  __attribute__((ext_vector_type(8)));
typedef __bf16 v8bf  __attribute__((ext_vector_type(8)));
typedef __bf16 v16bf __attribute__((ext_vector_type(16)));

// ----------------------------------------------------------- bf16 helpers
__device__ inline unsigned short f32_to_bf16_rne(float x) {
    unsigned u = __float_as_uint(x);
    unsigned r = u + 0x7FFFu + ((u >> 16) & 1u);
    return (unsigned short)(r >> 16);
}
__device__ inline float bf16_bits_to_f32(unsigned short h) {
    return __uint_as_float((unsigned)h << 16);
}

// ---------------------------------------------------------------- utilities
__global__ void zero_f32(float* __restrict__ p, long n) {
    long i = (long)blockIdx.x * blockDim.x + threadIdx.x;
    long stride = (long)gridDim.x * blockDim.x;
    for (; i < n; i += stride) p[i] = 0.0f;
}

__global__ void presence_scatter(const int* __restrict__ ids,
                                 float* __restrict__ presence, int n) {
    int i = blockIdx.x * blockDim.x + threadIdx.x;
    if (i < n) presence[ids[i]] = 1.0f;
}

__global__ void freq_recency(const float* __restrict__ presence,
                             const float* __restrict__ freq_in,
                             const float* __restrict__ rec_in,
                             float* __restrict__ freq_out,
                             float* __restrict__ rec_out) {
    int v = blockIdx.x * blockDim.x + threadIdx.x;
    if (v < VOCAB) {
        float p = presence[v];
        freq_out[v] = freq_in[v] + p;
        rec_out[v]  = (p > 0.0f) ? 1.0f : rec_in[v] * 0.99f;
    }
}

// ------------------------------------------------- bf16 hi/lo staging of X
// X[m][d] = token_emb[id][d] + fast_w[id][d] + pos_emb[m%SEQ][d]
__global__ void stage_split_X(const int* __restrict__ ids,
                              const float* __restrict__ token_emb,
                              const float* __restrict__ pos_emb,
                              const float* __restrict__ fast_w,
                              unsigned short* __restrict__ Xhi,
                              unsigned short* __restrict__ Xlo, int BS) {
    long t = (long)blockIdx.x * blockDim.x + threadIdx.x;
    long total = (long)BS * (EMBED / 4);
    if (t >= total) return;
    int m  = (int)(t / (EMBED / 4));
    int d4 = (int)(t - (long)m * (EMBED / 4)) * 4;
    int id = ids[m];
    int s  = m & (SEQ - 1);
    v4f a = *(const v4f*)(token_emb + (size_t)id * EMBED + d4);
    v4f b = *(const v4f*)(fast_w    + (size_t)id * EMBED + d4);
    v4f c = *(const v4f*)(pos_emb   + (size_t)s  * EMBED + d4);
    ushort4 hv, lv;
    float x0 = a.x + b.x + c.x, x1 = a.y + b.y + c.y;
    float x2 = a.z + b.z + c.z, x3 = a.w + b.w + c.w;
    hv.x = f32_to_bf16_rne(x0); lv.x = f32_to_bf16_rne(x0 - bf16_bits_to_f32(hv.x));
    hv.y = f32_to_bf16_rne(x1); lv.y = f32_to_bf16_rne(x1 - bf16_bits_to_f32(hv.y));
    hv.z = f32_to_bf16_rne(x2); lv.z = f32_to_bf16_rne(x2 - bf16_bits_to_f32(hv.z));
    hv.w = f32_to_bf16_rne(x3); lv.w = f32_to_bf16_rne(x3 - bf16_bits_to_f32(hv.w));
    *(ushort4*)(Xhi + (size_t)m * EMBED + d4) = hv;
    *(ushort4*)(Xlo + (size_t)m * EMBED + d4) = lv;
}

// ------------------------------------------------- bf16 hi/lo staging of W
__global__ void stage_split_W(const float* __restrict__ W,
                              unsigned short* __restrict__ Whi,
                              unsigned short* __restrict__ Wlo) {
    long t = (long)blockIdx.x * blockDim.x + threadIdx.x;
    long total = (long)EMBED * EMBED / 4;
    if (t >= total) return;
    long d4 = t * 4;
    v4f w = *(const v4f*)(W + d4);
    ushort4 hv, lv;
    hv.x = f32_to_bf16_rne(w.x); lv.x = f32_to_bf16_rne(w.x - bf16_bits_to_f32(hv.x));
    hv.y = f32_to_bf16_rne(w.y); lv.y = f32_to_bf16_rne(w.y - bf16_bits_to_f32(hv.y));
    hv.z = f32_to_bf16_rne(w.z); lv.z = f32_to_bf16_rne(w.z - bf16_bits_to_f32(hv.z));
    hv.w = f32_to_bf16_rne(w.w); lv.w = f32_to_bf16_rne(w.w - bf16_bits_to_f32(hv.w));
    *(ushort4*)(Whi + d4) = hv;
    *(ushort4*)(Wlo + d4) = lv;
}

// ------------------------------------------- per-position context accumulate
__global__ void ctx_accumulate(const int* __restrict__ ids,
                               const float* __restrict__ token_emb,
                               float* __restrict__ seg_sum,
                               float* __restrict__ seg_cnt) {
    int p    = blockIdx.x;
    int s    = p & (SEQ - 1);
    int row0 = p - s;
    int tid  = threadIdx.x;

    int   nbr[8];
    float cnt = 0.0f;
#pragma unroll
    for (int j = 0; j < 8; ++j) {
        int off = (j < 4) ? (j - 4) : (j - 3);
        int sp  = s + off;
        int id  = -1;
        if (sp >= 0 && sp < SEQ) {
            id = ids[row0 + sp];
            if (id < MIN_TOK) id = -1;
        }
        nbr[j] = id;
        cnt += (id >= 0) ? 1.0f : 0.0f;
    }
    float inv = 1.0f / fmaxf(cnt, 1.0f);

    float av[3];
    float sq = 0.0f;
#pragma unroll
    for (int r = 0; r < 3; ++r) {
        int d = tid + r * 256;
        float sum = 0.0f;
#pragma unroll
        for (int j = 0; j < 8; ++j)
            if (nbr[j] >= 0) sum += token_emb[(size_t)nbr[j] * EMBED + d];
        av[r] = sum * inv;
        sq += av[r] * av[r];
    }

    __shared__ float red[256];
    red[tid] = sq;
    __syncthreads();
    for (int w = 128; w > 0; w >>= 1) {
        if (tid < w) red[tid] += red[tid + w];
        __syncthreads();
    }
    float rnorm = 1.0f / fmaxf(sqrtf(red[0]), EPSF);

    int center = ids[p];
    if (center >= MIN_TOK && cnt > 0.0f) {
#pragma unroll
        for (int r = 0; r < 3; ++r) {
            int d = tid + r * 256;
            atomicAdd(&seg_sum[(size_t)center * EMBED + d], av[r] * rnorm);
        }
        if (tid == 0) atomicAdd(&seg_cnt[center], 1.0f);
    }
}

// ------------------------------------------------- per-vocab fast_w finalize
__global__ void finalize_fast(const float* __restrict__ token_emb,
                              const float* __restrict__ fast_w,
                              const float* __restrict__ seg_cnt,
                              const float* __restrict__ freq_new,
                              float* __restrict__ fastseg) {
    int v   = blockIdx.x;
    int tid = threadIdx.x;
    float cnt = seg_cnt[v];
    float inv = 1.0f / fmaxf(cnt, 1.0f);

    float avg[3];
    float a2 = 0.0f, t2 = 0.0f, dt = 0.0f;
#pragma unroll
    for (int r = 0; r < 3; ++r) {
        int d = tid + r * 256;
        float s = fastseg[(size_t)v * EMBED + d];
        float t = token_emb[(size_t)v * EMBED + d];
        avg[r] = s * inv;
        a2 += avg[r] * avg[r];
        t2 += t * t;
        dt += avg[r] * t;
    }
    __shared__ float ra[256], rt[256], rd[256];
    ra[tid] = a2; rt[tid] = t2; rd[tid] = dt;
    __syncthreads();
    for (int w = 128; w > 0; w >>= 1) {
        if (tid < w) { ra[tid] += ra[tid + w]; rt[tid] += rt[tid + w]; rd[tid] += rd[tid + w]; }
        __syncthreads();
    }
    float navg = fmaxf(sqrtf(ra[0]), EPSF);
    float nte  = fmaxf(sqrtf(rt[0]), EPSF);
    float sim  = rd[0] / (navg * nte);

    float rate = 0.01f * (1.0f - fminf(fmaxf(sim, 0.0f), 0.9f));
    float fq = freq_new[v];
    float ff = (fq > 100.0f) ? (1.0f / (1.0f + logf(fmaxf(fq, 1.0f) * 0.01f))) : 1.0f;
    rate *= ff;
    float has = (cnt > 0.0f) ? 1.0f : 0.0f;

#pragma unroll
    for (int r = 0; r < 3; ++r) {
        int d = tid + r * 256;
        float up = rate * (avg[r] / navg);
        up = fminf(fmaxf(up, -0.05f), 0.05f);
        fastseg[(size_t)v * EMBED + d] = fast_w[(size_t)v * EMBED + d] + up * has;
    }
}

// ---------------------------------------------- emb GEMM: 3xBF16 WMMA path
// per wave: 16(M) x 64(N), 4 f32 accumulators, K step 32 via
// v_wmma_f32_16x16x32_bf16; A = Xhi/Xlo, B(k,n) = W[n][k] hi/lo.
__global__ __launch_bounds__(256)
void emb_gemm_wmma_bf16(const unsigned short* __restrict__ Xhi_u,
                        const unsigned short* __restrict__ Xlo_u,
                        const unsigned short* __restrict__ Whi_u,
                        const unsigned short* __restrict__ Wlo_u,
                        const float* __restrict__ ctx_b,
                        float* __restrict__ emb_out,
                        int ntilesN) {
    const __bf16* Xhi = (const __bf16*)Xhi_u;
    const __bf16* Xlo = (const __bf16*)Xlo_u;
    const __bf16* Whi = (const __bf16*)Whi_u;
    const __bf16* Wlo = (const __bf16*)Wlo_u;

    int wid  = (blockIdx.x * blockDim.x + threadIdx.x) >> 5;
    int lane = threadIdx.x & 31;
    int mt = wid / ntilesN;
    int nt = wid - mt * ntilesN;
    int M0 = mt * 16;
    int N0 = nt * 64;
    int hl = lane >> 4;
    int lr = lane & 15;

    // A row pointers (16-bit A 16x32: lane holds contiguous K=[16*hl..+15])
    const __bf16* axh = Xhi + (size_t)(M0 + lr) * EMBED;
    const __bf16* axl = Xlo + (size_t)(M0 + lr) * EMBED;
    // B column pointers (16-bit B 32x16: lane holds contiguous K=[16*hl..+15] of col N0+16q+lr)
    const __bf16* bh[4];
    const __bf16* bl[4];
#pragma unroll
    for (int q = 0; q < 4; ++q) {
        bh[q] = Whi + (size_t)(N0 + 16 * q + lr) * EMBED;
        bl[q] = Wlo + (size_t)(N0 + 16 * q + lr) * EMBED;
    }

    v8f acc[4];
#pragma unroll
    for (int q = 0; q < 4; ++q) acc[q] = (v8f){};

    for (int kb = 0; kb < EMBED; kb += 32) {
        int ka = kb + 16 * hl;
        v8bf ah0 = *(const v8bf*)(axh + ka);
        v8bf ah1 = *(const v8bf*)(axh + ka + 8);
        v8bf al0 = *(const v8bf*)(axl + ka);
        v8bf al1 = *(const v8bf*)(axl + ka + 8);
        v16bf Ahi = __builtin_shufflevector(ah0, ah1, 0,1,2,3,4,5,6,7,8,9,10,11,12,13,14,15);
        v16bf Alo = __builtin_shufflevector(al0, al1, 0,1,2,3,4,5,6,7,8,9,10,11,12,13,14,15);

        int kbq = kb + 16 * hl;
        v16bf Bh[4], Bl[4];
#pragma unroll
        for (int q = 0; q < 4; ++q) {
            Bh[q] = *(const v16bf*)(bh[q] + kbq);
            Bl[q] = *(const v16bf*)(bl[q] + kbq);
        }
        // hi*hi, then hi*lo, then lo*hi: dependent acc updates 4 issues apart
#pragma unroll
        for (int q = 0; q < 4; ++q)
            acc[q] = __builtin_amdgcn_wmma_f32_16x16x32_bf16(false, Ahi, false, Bh[q],
                                                             (short)0, acc[q], false, false);
#pragma unroll
        for (int q = 0; q < 4; ++q)
            acc[q] = __builtin_amdgcn_wmma_f32_16x16x32_bf16(false, Ahi, false, Bl[q],
                                                             (short)0, acc[q], false, false);
#pragma unroll
        for (int q = 0; q < 4; ++q)
            acc[q] = __builtin_amdgcn_wmma_f32_16x16x32_bf16(false, Alo, false, Bh[q],
                                                             (short)0, acc[q], false, false);
    }

    // C/D: VGPR j, lane -> (M = j + 8*hl, N = lr) per 16-col subtile
#pragma unroll
    for (int j = 0; j < 8; ++j) {
        int row = M0 + j + 8 * hl;
        const unsigned short* xh = Xhi_u + (size_t)row * EMBED;
        const unsigned short* xl = Xlo_u + (size_t)row * EMBED;
#pragma unroll
        for (int q = 0; q < 4; ++q) {
            int c = N0 + 16 * q + lr;
            float x = bf16_bits_to_f32(xh[c]) + bf16_bits_to_f32(xl[c]);
            emb_out[(size_t)row * EMBED + c] = x + acc[q][j] + ctx_b[c];
        }
    }
}

// ------------------------------------- fallback: fp32 WMMA with A gather
__global__ __launch_bounds__(256)
void emb_gemm_wmma_f32(const int* __restrict__ ids,
                       const float* __restrict__ token_emb,
                       const float* __restrict__ pos_emb,
                       const float* __restrict__ fast_w,
                       const float* __restrict__ ctx_W,
                       const float* __restrict__ ctx_b,
                       float* __restrict__ emb_out,
                       int ntilesN) {
    int wid  = (blockIdx.x * blockDim.x + threadIdx.x) >> 5;
    int lane = threadIdx.x & 31;
    int mt = wid / ntilesN;
    int nt = wid - mt * ntilesN;
    int M0 = mt * 16;
    int N0 = nt * 64;
    int hl = lane >> 4;
    int lr = lane & 15;
    int koff = 2 * hl;

    const float* bp[4];
#pragma unroll
    for (int q = 0; q < 4; ++q)
        bp[q] = ctx_W + (size_t)(N0 + 16 * q + lr) * EMBED;

    v8f acc[4];
#pragma unroll
    for (int q = 0; q < 4; ++q) acc[q] = (v8f){};

    int am  = M0 + lr;
    int aid = ids[am];
    int as  = am & (SEQ - 1);
    const float* tokp = token_emb + (size_t)aid * EMBED;
    const float* fwp  = fast_w    + (size_t)aid * EMBED;
    const float* pep  = pos_emb   + (size_t)as  * EMBED;
#pragma unroll 2
    for (int k = 0; k < EMBED; k += 4) {
        int kk = k + koff;
        v2f t  = *(const v2f*)(tokp + kk);
        v2f f  = *(const v2f*)(fwp  + kk);
        v2f pv = *(const v2f*)(pep  + kk);
        v2f a;
        a.x = t.x + f.x + pv.x;
        a.y = t.y + f.y + pv.y;
#pragma unroll
        for (int q = 0; q < 4; ++q) {
            v2f b = *(const v2f*)(bp[q] + kk);
            acc[q] = __builtin_amdgcn_wmma_f32_16x16x4_f32(false, a, false, b,
                                                           (short)0, acc[q], false, false);
        }
    }

#pragma unroll
    for (int j = 0; j < 8; ++j) {
        int row = M0 + j + 8 * hl;
        int rid = ids[row];
        int rs  = row & (SEQ - 1);
        const float* tr = token_emb + (size_t)rid * EMBED;
        const float* fr = fast_w    + (size_t)rid * EMBED;
        const float* pr = pos_emb   + (size_t)rs  * EMBED;
#pragma unroll
        for (int q = 0; q < 4; ++q) {
            int c = N0 + 16 * q + lr;
            float x = tr[c] + fr[c] + pr[c];
            emb_out[(size_t)row * EMBED + c] = x + acc[q][j] + ctx_b[c];
        }
    }
}

// ----------------------------------------------------------------- launcher
extern "C" void kernel_launch(void* const* d_in, const int* in_sizes, int n_in,
                              void* d_out, int out_size, void* d_ws, size_t ws_size,
                              hipStream_t stream) {
    const int*   ids        = (const int*)  d_in[0];
    const float* token_emb  = (const float*)d_in[1];
    const float* pos_emb    = (const float*)d_in[2];
    const float* fast_w     = (const float*)d_in[3];
    const float* ctx_W      = (const float*)d_in[4];
    const float* ctx_b      = (const float*)d_in[5];
    const float* token_freq = (const float*)d_in[6];
    const float* token_rec  = (const float*)d_in[7];

    const int  BS   = in_sizes[0];            // B*S = 16384
    const long embN = (long)BS * EMBED;

    float* out      = (float*)d_out;
    float* emb_out  = out;
    float* fast_out = out + embN;                       // seg_sum scratch, then fast_new
    float* freq_out = fast_out + (long)VOCAB * EMBED;
    float* rec_out  = freq_out + VOCAB;

    float* presence = (float*)d_ws;
    float* seg_cnt  = presence + VOCAB;

    size_t base   = ((2UL * VOCAB * 4 + 63) / 64) * 64;   // bytes, 64B aligned
    size_t xBytes = (size_t)BS * EMBED * 2;
    size_t wBytes = (size_t)EMBED * EMBED * 2;
    unsigned short* Xhi = (unsigned short*)((char*)d_ws + base);
    unsigned short* Xlo = (unsigned short*)((char*)d_ws + base + xBytes);
    unsigned short* Whi = (unsigned short*)((char*)d_ws + base + 2 * xBytes);
    unsigned short* Wlo = (unsigned short*)((char*)d_ws + base + 2 * xBytes + wBytes);
    bool useBf16 = ws_size >= base + 2 * xBytes + 2 * wBytes;

    zero_f32<<<4096, 256, 0, stream>>>(fast_out, (long)VOCAB * EMBED);
    zero_f32<<<128,  256, 0, stream>>>(presence, 2L * VOCAB);

    presence_scatter<<<(BS + 255) / 256, 256, 0, stream>>>(ids, presence, BS);
    freq_recency<<<(VOCAB + 255) / 256, 256, 0, stream>>>(presence, token_freq, token_rec,
                                                          freq_out, rec_out);
    ctx_accumulate<<<BS, 256, 0, stream>>>(ids, token_emb, fast_out, seg_cnt);
    finalize_fast<<<VOCAB, 256, 0, stream>>>(token_emb, fast_w, seg_cnt, freq_out, fast_out);

    const int ntilesN = EMBED / 64;                      // 12
    const int waves   = (BS / 16) * ntilesN;             // multiple of 8
    if (useBf16) {
        long tx = (long)BS * (EMBED / 4);
        stage_split_X<<<(int)((tx + 255) / 256), 256, 0, stream>>>(ids, token_emb, pos_emb,
                                                                   fast_w, Xhi, Xlo, BS);
        long tw = (long)EMBED * EMBED / 4;
        stage_split_W<<<(int)((tw + 255) / 256), 256, 0, stream>>>(ctx_W, Whi, Wlo);
        emb_gemm_wmma_bf16<<<waves / 8, 256, 0, stream>>>(Xhi, Xlo, Whi, Wlo, ctx_b,
                                                          emb_out, ntilesN);
    } else {
        emb_gemm_wmma_f32<<<waves / 8, 256, 0, stream>>>(ids, token_emb, pos_emb, fast_w,
                                                         ctx_W, ctx_b, emb_out, ntilesN);
    }
}
